// GeometricNet_53850299957816
// MI455X (gfx1250) — matrix operand order repaired and checked by reading, not compiled, not run on hardware
//
#include <hip/hip_runtime.h>
#include <hip/hip_bf16.h>

#define NN_ (256*1024)
#define E_  (NN_*8)
#define D_  128
#define B_  256
#define N_  1024

typedef __attribute__((ext_vector_type(16))) _Float16 v16h;
typedef __attribute__((ext_vector_type(8)))  float    v8f;
typedef __attribute__((ext_vector_type(4)))  unsigned int u32x4;
typedef __attribute__((ext_vector_type(8)))  int      i32x8;
typedef __attribute__((ext_vector_type(4)))  int      i32x4;

// ---------------- init / utility ----------------
__global__ void k_init_masks(float* emask, unsigned char* nmask) {
  int i = blockIdx.x * blockDim.x + threadIdx.x;
  if (i < E_)  emask[i] = 1.0f;
  if (i < NN_) nmask[i] = 1;
}

__global__ void k_zero(float* p, size_t n) {
  size_t i = (size_t)blockIdx.x * blockDim.x + threadIdx.x;
  if (i < n) p[i] = 0.0f;
}

// x[node] = emb[x_ids[node]]   (float4 per thread)
__global__ void k_gather(const int* __restrict__ ids,
                         const float* __restrict__ emb,
                         float* __restrict__ x) {
  int t = blockIdx.x * blockDim.x + threadIdx.x;   // NN*32 threads
  int node = t >> 5;
  int g = (t & 31) << 2;
  int id = ids[node];
  float4 v = *(const float4*)(emb + (size_t)id * D_ + g);
  *(float4*)(x + (size_t)node * D_ + g) = v;
}

// Per-layer weight prep: Wt16[mat][n*128+k] = (f16) W[k*128+n]
__global__ void k_prep_w(const float* __restrict__ Wl,
                         const float* __restrict__ Wr,
                         _Float16* __restrict__ wt) {
  int i = blockIdx.x * blockDim.x + threadIdx.x;   // 0..32767
  int mat = i >> 14;
  int idx = i & 16383;
  int k = idx >> 7, n = idx & 127;
  const float* W = mat ? Wr : Wl;
  wt[mat * 16384 + n * 128 + k] = (_Float16)W[idx];
}

// segment-sum of masked messages + degree, via global float atomics
__global__ void k_aggregate(const float* __restrict__ x,
                            const int* __restrict__ src,
                            const int* __restrict__ dst,
                            const float* __restrict__ emask,
                            float* __restrict__ agg,
                            float* __restrict__ deg) {
  long long t = (long long)blockIdx.x * blockDim.x + threadIdx.x; // E*32
  int e = (int)(t >> 5);
  int g = (int)(t & 31) << 2;
  float m = emask[e];
  int d = dst[e];
  if (m != 0.0f) {
    int s = src[e];
    float4 v = *(const float4*)(x + (size_t)s * D_ + g);
    float* ap = agg + (size_t)d * D_ + g;
    atomicAdd(ap + 0, v.x * m);
    atomicAdd(ap + 1, v.y * m);
    atomicAdd(ap + 2, v.z * m);
    atomicAdd(ap + 3, v.w * m);
  }
  if (g == 0) atomicAdd(deg + d, m);
}

// ---- Tensor Data Mover: load a 128x128 f16 tile (row-major, contiguous)
// from global memory into LDS at byte offset lds_off. D# per ISA 8.3/8.4.
__device__ __forceinline__ void tdm_load_tile_f16(unsigned lds_off,
                                                  const _Float16* gsrc) {
  unsigned long long ga = (unsigned long long)(uintptr_t)gsrc;
  u32x4 g0 = {
    1u,                                   // count=1, user descriptor
    lds_off,                              // lds_addr (bytes)
    (unsigned)(ga & 0xFFFFFFFFull),       // global_addr[31:0]
    (unsigned)((ga >> 32) & 0x1FFFFFFull) // global_addr[56:32]
      | (2u << 30)                        // type=2 ("image")
  };
  i32x8 g1 = {
    0x00010000,          // workgroup_mask=0, data_size=1 (2 bytes)
    (int)(128u << 16),   // tensor_dim0 = 128 (bits 63:48)
    (int)(128u << 16),   // tensor_dim1 = 128 (bits 95:80)
    (int)(128u << 16),   // tile_dim0   = 128 (bits 127:112)
    128,                 // tile_dim1   = 128 (bits 143:128)
    128,                 // tensor_dim0_stride = 128 (bits 191:160)
    0, 0
  };
  i32x4 g2 = {0, 0, 0, 0};
  i32x4 g3 = {0, 0, 0, 0};
#if defined(__clang_major__) && (__clang_major__ >= 23)
  i32x8 z8 = {0, 0, 0, 0, 0, 0, 0, 0};
  __builtin_amdgcn_tensor_load_to_lds(g0, g1, g2, g3, z8, 0);
#else
  __builtin_amdgcn_tensor_load_to_lds(g0, g1, g2, g3, 0);
#endif
}

// ---------------- fused SAGE GEMM (WMMA f16 -> f32) ----------------
// y = relu( (agg/max(deg,1)) @ Wl + x @ Wr + bl )
// block = 128 threads (4 waves). Block computes 64 rows x 128 cols.
// Wave computes 16 rows x 128 cols = 8 WMMA 16x16 tiles, K-loop of 32.
// Pre-transposed f16 weights are TDM-loaded into LDS (2 x 32KB tiles).
__global__ void __launch_bounds__(128)
k_sage_gemm(const float* __restrict__ agg,
            const float* __restrict__ deg,
            const float* __restrict__ x,
            const _Float16* __restrict__ wt16,   // [2][128*128], n-major
            const float* __restrict__ bl,
            float* __restrict__ y) {
  __shared__ _Float16 sW[2][D_ * D_];   // 64 KB
  int tid = threadIdx.x;
  if (tid < 32) {   // wave 0 issues the tensor DMAs and waits on TENSORcnt
    tdm_load_tile_f16((unsigned)(uintptr_t)&sW[0][0], wt16);
    tdm_load_tile_f16((unsigned)(uintptr_t)&sW[1][0], wt16 + D_ * D_);
    __builtin_amdgcn_s_wait_tensorcnt(0);
  }
  __syncthreads();

  int wave = tid >> 5;
  int lane = tid & 31;
  int halfsel = lane >> 4;               // 0: lanes 0-15, 1: lanes 16-31
  int R = blockIdx.x * 64 + wave * 16;   // wave's row base
  int lrow = R + (lane & 15);            // row this lane supplies to A

  v8f acc[8];
  #pragma unroll
  for (int nt = 0; nt < 8; ++nt)
    #pragma unroll
    for (int r = 0; r < 8; ++r) acc[nt][r] = 0.0f;

  #pragma unroll
  for (int mat = 0; mat < 2; ++mat) {
    const float* srcm = mat ? x : agg;
    float scale = mat ? 1.0f : 1.0f / fmaxf(deg[lrow], 1.0f);
    const float* arow = srcm + (size_t)lrow * D_;
    const _Float16* wt = sW[mat];
    #pragma unroll
    for (int kb = 0; kb < D_; kb += 32) {
      // A fragment (16x32 f16), contiguous layout (matches B):
      // lanes 0-15 : elems 0..15 = K kb+0..15
      // lanes 16-31: elems 0..15 = K kb+16..31
      v16h a;
      int k0 = kb + (halfsel << 4);
      #pragma unroll
      for (int j = 0; j < 8; ++j) a[j]     = (_Float16)(arow[k0 + j] * scale);
      #pragma unroll
      for (int j = 0; j < 8; ++j) a[8 + j] = (_Float16)(arow[k0 + 8 + j] * scale);

      // Prefetch all 8 B fragments (one DS clause), then chain 8 WMMAs
      v16h bf[8];
      #pragma unroll
      for (int nt = 0; nt < 8; ++nt) {
        int col = nt * 16 + (lane & 15);
        const _Float16* wrow = wt + col * D_ + kb + (halfsel << 4);
        #pragma unroll
        for (int j = 0; j < 16; ++j) bf[nt][j] = wrow[j];
      }
      #pragma unroll
      for (int nt = 0; nt < 8; ++nt) {
        acc[nt] = __builtin_amdgcn_wmma_f32_16x16x32_f16(
            false, a, false, bf[nt], (short)0, acc[nt], false, false);
      }
    }
  }

  // C/D layout: lane 0-15 col=N, vgpr r = row r; lane 16-31 rows 8..15
  int rbase = R + (halfsel << 3);
  #pragma unroll
  for (int nt = 0; nt < 8; ++nt) {
    int col = nt * 16 + (lane & 15);
    float bb = bl[col];
    #pragma unroll
    for (int r = 0; r < 8; ++r) {
      float v = acc[nt][r] + bb;
      y[(size_t)(rbase + r) * D_ + col] = v > 0.0f ? v : 0.0f;
    }
  }
}

// ---------------- topk pooling ----------------
__global__ void k_score(const float* __restrict__ x,
                        const float* __restrict__ p,
                        const unsigned char* __restrict__ nmask,
                        float* __restrict__ score) {
  int i = blockIdx.x * blockDim.x + threadIdx.x;
  if (i >= NN_) return;
  float pn = 0.0f, dt = 0.0f;
  const float* xr = x + (size_t)i * D_;
  for (int j = 0; j < D_; ++j) { float pv = p[j]; pn += pv * pv; dt += xr[j] * pv; }
  score[i] = nmask[i] ? dt * rsqrtf(pn) : -__builtin_inff();
}

// rank = #{s_j > s_i} + #{s_j == s_i, j < i}; keep if rank < k and unmasked
__global__ void __launch_bounds__(256)
k_topk(const float* __restrict__ score,
       unsigned char* __restrict__ nmask,
       float* __restrict__ tfac, int k) {
  __shared__ float ss[N_];
  int b = blockIdx.x;
  const float* sg = score + (size_t)b * N_;
  for (int i = threadIdx.x; i < N_; i += 256) ss[i] = sg[i];
  __syncthreads();
  for (int i = threadIdx.x; i < N_; i += 256) {
    float si = ss[i];
    int cnt = 0;
    for (int j = 0; j < N_; ++j) {
      float sj = ss[j];
      cnt += (sj > si) || (sj == si && j < i);
    }
    int gi = b * N_ + i;
    bool keep = (cnt < k) && nmask[gi] && (si != -__builtin_inff());
    nmask[gi] = keep ? 1 : 0;
    tfac[gi] = keep ? tanhf(si) : 0.0f;
  }
}

// x = keep ? x * tanh(score) : 0   (tfac already folds both)
__global__ void k_apply(float* __restrict__ x, const float* __restrict__ tfac) {
  int t = blockIdx.x * blockDim.x + threadIdx.x;  // NN*32
  int node = t >> 5;
  int g = (t & 31) << 2;
  float f = tfac[node];
  float4 v = *(float4*)(x + (size_t)node * D_ + g);
  v.x *= f; v.y *= f; v.z *= f; v.w *= f;
  *(float4*)(x + (size_t)node * D_ + g) = v;
}

__global__ void k_emask(float* __restrict__ emask,
                        const unsigned char* __restrict__ nmask,
                        const int* __restrict__ src,
                        const int* __restrict__ dst) {
  int e = blockIdx.x * blockDim.x + threadIdx.x;
  if (e >= E_) return;
  if (!(nmask[src[e]] && nmask[dst[e]])) emask[e] = 0.0f;
}

// h[b] += concat(max, mean) over kept nodes of graph b
__global__ void __launch_bounds__(128)
k_readout(const float* __restrict__ x,
          const unsigned char* __restrict__ nmask,
          float* __restrict__ h) {
  int b = blockIdx.x, d = threadIdx.x;
  float sum = 0.0f, mx = -__builtin_inff();
  int cnt = 0;
  const unsigned char* nm = nmask + (size_t)b * N_;
  const float* xb = x + (size_t)b * N_ * D_;
  for (int n = 0; n < N_; ++n) {
    if (nm[n]) {
      float v = xb[(size_t)n * D_ + d];
      sum += v;
      if (v > mx) mx = v;
      cnt++;
    }
  }
  float c = (float)(cnt > 0 ? cnt : 1);
  h[(size_t)b * 256 + d]       += mx;
  h[(size_t)b * 256 + 128 + d] += sum / c;
}

// ---------------- MLP head ----------------
__global__ void __launch_bounds__(128)
k_mlp(const float* __restrict__ h,
      const float* __restrict__ W1, const float* __restrict__ b1,
      const float* __restrict__ W2, const float* __restrict__ b2,
      const float* __restrict__ W3, const float* __restrict__ b3,
      float* __restrict__ out) {
  __shared__ float hr[256];
  __shared__ float t1[128];
  __shared__ float t2[64];
  int b = blockIdx.x, t = threadIdx.x;
  hr[t]       = h[(size_t)b * 256 + t];
  hr[t + 128] = h[(size_t)b * 256 + t + 128];
  __syncthreads();
  float s = b1[t];
  for (int j = 0; j < 256; ++j) s += hr[j] * W1[j * 128 + t];
  t1[t] = s > 0.0f ? s : 0.0f;
  __syncthreads();
  if (t < 64) {
    float s2 = b2[t];
    for (int j = 0; j < 128; ++j) s2 += t1[j] * W2[j * 64 + t];
    t2[t] = s2 > 0.0f ? s2 : 0.0f;
  }
  __syncthreads();
  if (t == 0) {
    float s3 = b3[0];
    for (int j = 0; j < 64; ++j) s3 += t2[j] * W3[j];
    out[b] = 1.0f / (1.0f + expf(-s3));
  }
}

// ---------------- launch ----------------
extern "C" void kernel_launch(void* const* d_in, const int* in_sizes, int n_in,
                              void* d_out, int out_size, void* d_ws, size_t ws_size,
                              hipStream_t stream) {
  (void)in_sizes; (void)n_in; (void)out_size; (void)ws_size;
  const int*   x_ids = (const int*)d_in[0];
  const int*   eidx  = (const int*)d_in[1];
  const int*   src   = eidx;
  const int*   dst   = eidx + E_;
  const float* emb   = (const float*)d_in[3];
  const float* Wl[3] = {(const float*)d_in[4],  (const float*)d_in[8],  (const float*)d_in[12]};
  const float* bl[3] = {(const float*)d_in[5],  (const float*)d_in[9],  (const float*)d_in[13]};
  const float* Wr[3] = {(const float*)d_in[6],  (const float*)d_in[10], (const float*)d_in[14]};
  const float* pv[3] = {(const float*)d_in[7],  (const float*)d_in[11], (const float*)d_in[15]};
  const float* l1W = (const float*)d_in[16];
  const float* l1b = (const float*)d_in[17];
  const float* l2W = (const float*)d_in[18];
  const float* l2b = (const float*)d_in[19];
  const float* l3W = (const float*)d_in[20];
  const float* l3b = (const float*)d_in[21];
  float* out = (float*)d_out;

  char* ws = (char*)d_ws;
  size_t off = 0;
  float* X   = (float*)(ws + off); off += (size_t)NN_ * D_ * 4;
  float* Y   = (float*)(ws + off); off += (size_t)NN_ * D_ * 4;
  float* AG  = (float*)(ws + off); off += (size_t)NN_ * D_ * 4;
  float* DEG = (float*)(ws + off); off += (size_t)NN_ * 4;   // contiguous after AG
  float* SC  = (float*)(ws + off); off += (size_t)NN_ * 4;
  float* TF  = (float*)(ws + off); off += (size_t)NN_ * 4;
  float* EM  = (float*)(ws + off); off += (size_t)E_ * 4;
  float* H   = (float*)(ws + off); off += (size_t)B_ * 256 * 4;
  unsigned char* NM = (unsigned char*)(ws + off); off += NN_;
  off = (off + 255) & ~(size_t)255;
  _Float16* WT = (_Float16*)(ws + off); off += (size_t)3 * 2 * D_ * D_ * 2;

  const int kvals[3] = {820, 656, 525};   // ceil(0.8*N), then 0.8 of that, etc.

  k_init_masks<<<E_ / 256, 256, 0, stream>>>(EM, NM);
  k_gather<<<(NN_ * 32) / 256, 256, 0, stream>>>(x_ids, emb, X);
  k_zero<<<(B_ * 256) / 256, 256, 0, stream>>>(H, (size_t)B_ * 256);
  for (int L = 0; L < 3; ++L)
    k_prep_w<<<(2 * D_ * D_) / 256, 256, 0, stream>>>(Wl[L], Wr[L],
                                                      WT + (size_t)L * 2 * D_ * D_);

  float* cur = X;
  float* nxt = Y;
  for (int L = 0; L < 3; ++L) {
    size_t agn = (size_t)NN_ * (D_ + 1);  // AG + DEG contiguous
    k_zero<<<(unsigned)((agn + 255) / 256), 256, 0, stream>>>(AG, agn);
    k_aggregate<<<(E_ * 32) / 256, 256, 0, stream>>>(cur, src, dst, EM, AG, DEG);
    k_sage_gemm<<<NN_ / 64, 128, 0, stream>>>(AG, DEG, cur,
                                              WT + (size_t)L * 2 * D_ * D_,
                                              bl[L], nxt);
    k_score<<<NN_ / 256, 256, 0, stream>>>(nxt, pv[L], NM, SC);
    k_topk<<<B_, 256, 0, stream>>>(SC, NM, TF, kvals[L]);
    k_apply<<<(NN_ * 32) / 256, 256, 0, stream>>>(nxt, TF);
    k_emask<<<E_ / 256, 256, 0, stream>>>(EM, NM, src, dst);
    k_readout<<<B_, 128, 0, stream>>>(nxt, NM, H);
    float* tmp = cur; cur = nxt; nxt = tmp;
  }

  k_mlp<<<B_, 128, 0, stream>>>(H, l1W, l1b, l2W, l2b, l3W, l3b, out);
}
